// BaseGumbelGraphNetwork_81621558493432
// MI455X (gfx1250) — matrix operand order repaired and checked by reading, compiled, not run on hardware
//
#include <hip/hip_runtime.h>

typedef __attribute__((ext_vector_type(16))) _Float16 v16h;
typedef __attribute__((ext_vector_type(8)))  float    v8f;

union Frag16 { v16h v; unsigned u[8]; _Float16 h[16]; };

#define Nn 512
#define Hh 64
#define NW 8   // waves per workgroup (wave32)

__launch_bounds__(256, 2)
__global__ void ggn_fused(const float* __restrict__ input,
                          const float* __restrict__ adj,
                          const float* __restrict__ W_n2e, const float* __restrict__ b_n2e,
                          const float* __restrict__ W_e2e, const float* __restrict__ b_e2e,
                          const float* __restrict__ W_e2n, const float* __restrict__ b_e2n,
                          const float* __restrict__ W_n2n, const float* __restrict__ b_n2n,
                          const float* __restrict__ W_o1,  const float* __restrict__ b_o1,
                          const float* __restrict__ W_o2,  const float* __restrict__ b_o2,
                          float* __restrict__ out)
{
    __shared__ __align__(16) _Float16 e1[NW][16][64];   // per-wave e1 tile (j x h_in), f16
    __shared__ __align__(16) float s_adj[Nn];           // adj row for this i
    __shared__ __align__(16) float s_x[Nn * 2];         // x[b, :, 0:2]
    __shared__ __align__(16) float s_part[NW][Hh];      // per-wave partial s
    __shared__ float s_v0[Hh];
    __shared__ float s_v1[Hh];

    const int tid  = threadIdx.x;
    const int wave = tid >> 5;
    const int lane = tid & 31;
    const int wg   = blockIdx.x;
    const int bi   = wg >> 9;     // b
    const int ii   = wg & 511;    // i

    // ---- stage adj row and all x_j for this batch into LDS via the CDNA5
    //      async memory->LDS path (tracked with ASYNCcnt, no VGPR round trip) ----
    if (tid < 128) {
        unsigned lds_a = (unsigned)(size_t)&s_adj[tid * 4];
        const float* g_a = &adj[ii * Nn + tid * 4];
        asm volatile("global_load_async_to_lds_b128 %0, %1, off"
                     :: "v"(lds_a), "v"(g_a) : "memory");
    }
    {
        unsigned lds_x = (unsigned)(size_t)&s_x[tid * 4];
        const float* g_x = &input[bi * Nn * 2 + tid * 4];
        asm volatile("global_load_async_to_lds_b128 %0, %1, off"
                     :: "v"(lds_x), "v"(g_x) : "memory");
    }

    const float xi0 = input[(bi * Nn + ii) * 2 + 0];
    const float xi1 = input[(bi * Nn + ii) * 2 + 1];

    // ---- per-lane n2e constants: lane handles h = lane and h = lane+32 ----
    // e1[j,h] = relu( b_n2e[h] + xi0*W[h,0] + xi1*W[h,1] + xj0*W[h,2] + xj1*W[h,3] )
    const float* wrL = &W_n2e[lane * 4];
    const float* wrH = &W_n2e[(lane + 32) * 4];
    const float ci_lo = b_n2e[lane]      + xi0 * wrL[0] + xi1 * wrL[1];
    const float w2_lo = wrL[2], w3_lo = wrL[3];
    const float ci_hi = b_n2e[lane + 32] + xi0 * wrH[0] + xi1 * wrH[1];
    const float w2_hi = wrH[2], w3_hi = wrH[3];

    // ---- B fragments for W_e2e (constant for whole kernel) ----
    // B[k=h_in][n=h_out] = W_e2e[n][k]; lane%16 <-> N, halves 2v,2v+1 <-> K in table order (matches A)
    Frag16 bf[2][4];
    #pragma unroll
    for (int kc = 0; kc < 2; ++kc)
        #pragma unroll
        for (int nt = 0; nt < 4; ++nt)
            #pragma unroll
            for (int v = 0; v < 8; ++v) {
                const int n0 = nt * 16 + (lane & 15);
                const int kk = kc * 32 + ((v < 4) ? (2 * v + 8 * (lane >> 4)) : (16 + 2 * (v - 4) + 8 * (lane >> 4)));
                bf[kc][nt].h[2 * v]     = (_Float16)W_e2e[n0 * Hh + kk];
                bf[kc][nt].h[2 * v + 1] = (_Float16)W_e2e[n0 * Hh + kk + 1];
            }
    float bee[4];
    #pragma unroll
    for (int nt = 0; nt < 4; ++nt) bee[nt] = b_e2e[nt * 16 + (lane & 15)];

    // async LDS fills must land before any wave touches s_adj/s_x
    asm volatile("s_wait_asynccnt 0" ::: "memory");
    __syncthreads();

    // ---- main loop: wave owns j in [wave*64, wave*64+64), 4 tiles of 16 ----
    float acc[4] = {0.f, 0.f, 0.f, 0.f};
    const int jwbase = wave * 64;
    const int m  = lane & 15;
    const int hs = lane >> 4;

    for (int jt = 0; jt < 4; ++jt) {
        const int jb = jwbase + jt * 16;

        // build e1 tile: lane writes h=lane and h=lane+32 for all 16 j rows
        #pragma unroll
        for (int jj = 0; jj < 16; ++jj) {
            const int j = jb + jj;
            const float xj0 = s_x[2 * j], xj1 = s_x[2 * j + 1];
            float lo = ci_lo + xj0 * w2_lo + xj1 * w3_lo; lo = lo > 0.f ? lo : 0.f;
            float hi = ci_hi + xj0 * w2_hi + xj1 * w3_hi; hi = hi > 0.f ? hi : 0.f;
            e1[wave][jj][lane]      = (_Float16)lo;
            e1[wave][jj][lane + 32] = (_Float16)hi;
        }
        // wave-local producer->consumer across lanes through LDS
        asm volatile("s_wait_dscnt 0" ::: "memory");

        // A fragments per ISA 16-bit A 16x32 layout (contiguous per lane -> ds_load_b128)
        Frag16 a0, a1;
        #pragma unroll
        for (int v = 0; v < 8; ++v) {
            const int kk = (v < 4) ? (2 * v + 8 * hs) : (16 + 2 * (v - 4) + 8 * hs);
            a0.u[v] = *(const unsigned*)&e1[wave][m][kk];
            a1.u[v] = *(const unsigned*)&e1[wave][m][32 + kk];
        }
        // adjacency weights for the 8 M-rows this lane's accumulators cover
        float aw[8];
        #pragma unroll
        for (int r = 0; r < 8; ++r) aw[r] = s_adj[jb + r + 8 * hs];

        // e2 = relu(e1 @ W_e2e^T + b); s += adj * e2, reduced over M in-register
        #pragma unroll
        for (int nt = 0; nt < 4; ++nt) {
            v8f c = {};
            c = __builtin_amdgcn_wmma_f32_16x16x32_f16(false, a0.v, false, bf[0][nt].v,
                                                       (short)0, c, false, false);
            c = __builtin_amdgcn_wmma_f32_16x16x32_f16(false, a1.v, false, bf[1][nt].v,
                                                       (short)0, c, false, false);
            float p = 0.f;
            #pragma unroll
            for (int r = 0; r < 8; ++r) {
                float e2 = c[r] + bee[nt];
                e2 = e2 > 0.f ? e2 : 0.f;
                p += e2 * aw[r];
            }
            acc[nt] += p;
        }
    }

    // fold the two half-wave M groups: lanes L and L+16 hold same N = L%16
    #pragma unroll
    for (int nt = 0; nt < 4; ++nt) acc[nt] += __shfl_xor(acc[nt], 16, 32);
    if (lane < 16) {
        #pragma unroll
        for (int nt = 0; nt < 4; ++nt) s_part[wave][nt * 16 + lane] = acc[nt];
    }
    __syncthreads();

    // ---- tail: s -> e2n -> n2n -> concat -> o1 -> o2 (tiny, VALU) ----
    if (tid < Hh) {
        float s = 0.f;
        #pragma unroll
        for (int w = 0; w < NW; ++w) s += s_part[w][tid];
        s_v0[tid] = s;
    }
    __syncthreads();
    if (tid < Hh) {
        float a = b_e2n[tid];
        for (int k = 0; k < Hh; ++k) a += s_v0[k] * W_e2n[tid * Hh + k];
        s_v1[tid] = a > 0.f ? a : 0.f;
    }
    __syncthreads();
    if (tid < Hh) {
        float a = b_n2n[tid];
        for (int k = 0; k < Hh; ++k) a += s_v1[k] * W_n2n[tid * Hh + k];
        s_v0[tid] = a > 0.f ? a : 0.f;
    }
    __syncthreads();
    if (tid < Hh) {
        const float* w = &W_o1[tid * 66];
        float a = b_o1[tid] + xi0 * w[0] + xi1 * w[1];
        for (int k = 0; k < Hh; ++k) a += s_v0[k] * w[2 + k];
        s_v1[tid] = a > 0.f ? a : 0.f;
    }
    __syncthreads();
    if (tid < 2) {
        float a = b_o2[tid];
        for (int k = 0; k < Hh; ++k) a += s_v1[k] * W_o2[tid * Hh + k];
        out[wg * 2 + tid] = a;
    }
}

extern "C" void kernel_launch(void* const* d_in, const int* in_sizes, int n_in,
                              void* d_out, int out_size, void* d_ws, size_t ws_size,
                              hipStream_t stream)
{
    const float* input = (const float*)d_in[0];
    const float* adj   = (const float*)d_in[1];
    const float* W_n2e = (const float*)d_in[2];
    const float* b_n2e = (const float*)d_in[3];
    const float* W_e2e = (const float*)d_in[4];
    const float* b_e2e = (const float*)d_in[5];
    const float* W_e2n = (const float*)d_in[6];
    const float* b_e2n = (const float*)d_in[7];
    const float* W_n2n = (const float*)d_in[8];
    const float* b_n2n = (const float*)d_in[9];
    const float* W_o1  = (const float*)d_in[10];
    const float* b_o1  = (const float*)d_in[11];
    const float* W_o2  = (const float*)d_in[12];
    const float* b_o2  = (const float*)d_in[13];
    float* out = (float*)d_out;

    // one workgroup per (b, i): 4*512 = 2048 blocks, 8 waves each
    ggn_fused<<<2048, 256, 0, stream>>>(input, adj,
                                        W_n2e, b_n2e, W_e2e, b_e2e,
                                        W_e2n, b_e2n, W_n2n, b_n2n,
                                        W_o1, b_o1, W_o2, b_o2, out);
}
